// GraphTransformerLayer_29824252903964
// MI455X (gfx1250) — compile-verified, matrix operand order checked by reading, not run on hardware
//
#include <hip/hip_runtime.h>
#include <hip/hip_bf16.h>
#include <math.h>

// Graph Transformer layer for MI455X (gfx1250), wave32 + WMMA bf16.
#define D_   128
#define H_   8
#define DH_  16
#define DFF_ 256
#define N_   50000
#define E_   800000

typedef __bf16 bf16;
typedef __attribute__((ext_vector_type(16))) __bf16 v16bf;
typedef __attribute__((ext_vector_type(8)))  float  v8f;

// ---------------- WMMA fragment loaders (16x16x32 bf16) ----------------
// A fragment: rows [mrow,mrow+16), K-cols [kb,kb+32), row-major fp32 source.
// ISA layout: lane l<16 -> row mrow+l holds K = kb+{0..7,16..23};
//             lane l>=16 -> row mrow+(l-16) holds K = kb+{8..15,24..31}.
__device__ inline v16bf load_a_frag(const float* __restrict__ A, int lda,
                                    int mrow, int kb, int lane) {
  int hf = lane >> 4;
  const float* row = A + (size_t)(mrow + (lane & 15)) * lda + kb + hf * 8;
  v16bf a;
#pragma unroll
  for (int i = 0; i < 8; ++i) a[i] = (bf16)row[i];
#pragma unroll
  for (int i = 0; i < 8; ++i) a[8 + i] = (bf16)row[16 + i];
  return a;
}

// Same, but A element = wV[m][k] / (z[m][k/16] + 1e-6)  (attention normalize)
__device__ inline v16bf load_a_frag_norm(const float* __restrict__ A, int lda,
                                         int mrow, int kb, int lane,
                                         const float* __restrict__ z) {
  int hf = lane >> 4;
  int m  = mrow + (lane & 15);
  const float* row = A + (size_t)m * lda + kb + hf * 8;
  const float* zr  = z + (size_t)m * H_;
  int k0 = kb + hf * 8;
  v16bf a;
#pragma unroll
  for (int i = 0; i < 8; ++i) {
    float inv = 1.0f / (zr[(k0 + i) >> 4] + 1e-6f);
    a[i] = (bf16)(row[i] * inv);
  }
#pragma unroll
  for (int i = 0; i < 8; ++i) {
    float inv = 1.0f / (zr[(k0 + 16 + i) >> 4] + 1e-6f);
    a[8 + i] = (bf16)(row[16 + i] * inv);
  }
  return a;
}

// B fragment: K-rows [kb,kb+32), cols [ncol,ncol+16), row-major fp32 [K x M].
__device__ inline v16bf load_b_frag(const float* __restrict__ B, int ldb,
                                    int kb, int ncol, int lane) {
  int hf = lane >> 4;
  const float* col = B + (size_t)(kb + hf * 8) * ldb + ncol + (lane & 15);
  v16bf b;
#pragma unroll
  for (int i = 0; i < 8; ++i) b[i] = (bf16)col[(size_t)i * ldb];
#pragma unroll
  for (int i = 0; i < 8; ++i) b[8 + i] = (bf16)col[(size_t)(16 + i) * ldb];
  return b;
}

// ---------------- Generic WMMA GEMM with fused epilogues ----------------
// MODE 0: C = A@B                                  (QKV projections)
// MODE 1: A normalized by z; C = A@B + bias + res; accumulate BN stats
// MODE 2: C = relu(A@B + bias)                     (FFN hidden)
// MODE 3: C = A@B + bias + res; accumulate BN stats (FFN out)
template <int MODE>
__global__ void gemm_wmma(const float* __restrict__ A, int lda,
                          const float* __restrict__ B, int ldb, int Kdim,
                          float* __restrict__ C, int ldc,
                          const float* __restrict__ bias,
                          const float* __restrict__ resid,
                          const float* __restrict__ zbuf,
                          float* __restrict__ statSum,
                          float* __restrict__ statSq) {
  int lane = threadIdx.x & 31;
  int wave = threadIdx.x >> 5;
  int mrow = blockIdx.x * 16;
  int ncol = (blockIdx.y * 8 + wave) * 16;

  v8f acc = {};
  for (int kb = 0; kb < Kdim; kb += 32) {
    v16bf a = (MODE == 1) ? load_a_frag_norm(A, lda, mrow, kb, lane, zbuf)
                          : load_a_frag(A, lda, mrow, kb, lane);
    v16bf b = load_b_frag(B, ldb, kb, ncol, lane);
    acc = __builtin_amdgcn_wmma_f32_16x16x32_bf16(
        /*neg_a=*/false, a, /*neg_b=*/false, b,
        /*c_mod=*/(short)0, acc, /*reuse_a=*/false, /*reuse_b=*/false);
  }

  // C/D layout: lane l<16 -> col ncol+l, rows mrow+r (VGPR r);
  //             lane l>=16 -> same col, rows mrow+8+r.
  int n     = ncol + (lane & 15);
  int rbase = mrow + (lane >> 4) * 8;
  float bia = (MODE == 0) ? 0.0f : bias[n];
  float ps = 0.0f, psq = 0.0f;
#pragma unroll
  for (int r = 0; r < 8; ++r) {
    float c = acc[r] + bia;
    if (MODE == 1 || MODE == 3) c += resid[(size_t)(rbase + r) * D_ + n];
    if (MODE == 2) c = fmaxf(c, 0.0f);
    C[(size_t)(rbase + r) * ldc + n] = c;
    if (MODE == 1 || MODE == 3) { ps += c; psq += c * c; }
  }
  if (MODE == 1 || MODE == 3) {
    atomicAdd(statSum + n, ps);
    atomicAdd(statSq  + n, psq);
  }
}

// ---------------- Edge attention: one thread per (edge, head) -----------
__global__ void edge_attn(const float* __restrict__ Q, const float* __restrict__ K,
                          const float* __restrict__ V,
                          const int* __restrict__ src, const int* __restrict__ dst,
                          float* __restrict__ wV, float* __restrict__ z, int nE) {
  int t = blockIdx.x * blockDim.x + threadIdx.x;
  if (t >= nE * H_) return;
  int e  = t >> 3;
  int hd = t & 7;
  int s = src[e];
  int d = dst[e];
  const float4* kp = (const float4*)(K + (size_t)s * D_ + hd * DH_);
  const float4* qp = (const float4*)(Q + (size_t)d * D_ + hd * DH_);
  float dot = 0.0f;
#pragma unroll
  for (int i = 0; i < 4; ++i) {
    float4 kv = kp[i];
    float4 qv = qp[i];
    dot += kv.x * qv.x + kv.y * qv.y + kv.z * qv.z + kv.w * qv.w;
  }
  dot *= 0.25f;                           // 1/sqrt(DH)
  dot = fminf(5.0f, fmaxf(-5.0f, dot));
  float sv = expf(dot);
  const float4* vp = (const float4*)(V + (size_t)s * D_ + hd * DH_);
  float* outp = wV + (size_t)d * D_ + hd * DH_;
#pragma unroll
  for (int i = 0; i < 4; ++i) {
    float4 vv = vp[i];
    atomicAdd(outp + 4 * i + 0, vv.x * sv);
    atomicAdd(outp + 4 * i + 1, vv.y * sv);
    atomicAdd(outp + 4 * i + 2, vv.z * sv);
    atomicAdd(outp + 4 * i + 3, vv.w * sv);
  }
  atomicAdd(z + (size_t)d * H_ + hd, sv);
}

// ---------------- BatchNorm helpers ----------------
__global__ void bn_finalize(const float* __restrict__ sum, const float* __restrict__ sq,
                            float* __restrict__ mu, float* __restrict__ rstd, float invN) {
  int c = threadIdx.x;            // 128 threads
  float m = sum[c] * invN;
  float v = sq[c] * invN - m * m;
  mu[c]   = m;
  rstd[c] = rsqrtf(v + 1e-5f);
}

__global__ void bn_apply(const float* __restrict__ x, const float* __restrict__ mu,
                         const float* __restrict__ rstd, const float* __restrict__ g,
                         const float* __restrict__ b, float* __restrict__ out, int total) {
  int i = blockIdx.x * blockDim.x + threadIdx.x;
  if (i >= total) return;
  int c = i & (D_ - 1);
  out[i] = (x[i] - mu[c]) * rstd[c] * g[c] + b[c];
}

// ---------------- Host launch ----------------
extern "C" void kernel_launch(void* const* d_in, const int* in_sizes, int n_in,
                              void* d_out, int out_size, void* d_ws, size_t ws_size,
                              hipStream_t stream) {
  const float* h   = (const float*)d_in[0];
  const int*   src = (const int*)d_in[1];
  const int*   dst = (const int*)d_in[2];
  const float* Wq  = (const float*)d_in[3];
  const float* Wk  = (const float*)d_in[4];
  const float* Wv  = (const float*)d_in[5];
  const float* Wo  = (const float*)d_in[6];
  const float* bo  = (const float*)d_in[7];
  const float* W1  = (const float*)d_in[8];
  const float* b1  = (const float*)d_in[9];
  const float* W2  = (const float*)d_in[10];
  const float* b2  = (const float*)d_in[11];
  const float* g1  = (const float*)d_in[12];
  const float* be1 = (const float*)d_in[13];
  const float* g2  = (const float*)d_in[14];
  const float* be2 = (const float*)d_in[15];

  float* ws = (float*)d_ws;
  const size_t ND = (size_t)N_ * D_;
  float* Qb   = ws;                              // [N,128]
  float* Kb   = ws + ND;                         // [N,128]
  float* Vb   = ws + 2 * ND;                     // [N,128]
  float* WVb  = ws + 3 * ND;                     // [N,128] scatter target
  float* Zb   = ws + 4 * ND;                     // [N,8]   scatter target
  float* st   = ws + 4 * ND + (size_t)N_ * H_;   // stats: 8 x 128 floats
  float* sum1 = st;        float* sq1 = st + 128;
  float* mu1  = st + 256;  float* rs1 = st + 384;
  float* sum2 = st + 512;  float* sq2 = st + 640;
  float* mu2  = st + 768;  float* rs2 = st + 896;
  // Aliased buffers (lifetimes are disjoint):
  float* X1 = Qb;   // pre-BN1  (Q dead after edge kernel)
  float* HX = Kb;   // post-BN1 (K dead after edge kernel)
  float* Tb = Vb;   // FFN hidden [N,256] spans Vb..WVb (both dead then)
  float* Yb = Qb;   // pre-BN2  (X1 dead after BN1 apply)
  float* outp = (float*)d_out;

  // Zero scatter targets + stat accumulators (one contiguous region).
  hipMemsetAsync(WVb, 0, (ND + (size_t)N_ * H_ + 1024) * sizeof(float), stream);

  dim3 blk(256);
  dim3 g128(N_ / 16, 1);   // 128 output cols: 8 waves/block cover them
  dim3 g256(N_ / 16, 2);   // 256 output cols

  // 1) Q/K/V projections
  gemm_wmma<0><<<g128, blk, 0, stream>>>(h, D_, Wq, D_, D_, Qb, D_, nullptr, nullptr, nullptr, nullptr, nullptr);
  gemm_wmma<0><<<g128, blk, 0, stream>>>(h, D_, Wk, D_, D_, Kb, D_, nullptr, nullptr, nullptr, nullptr, nullptr);
  gemm_wmma<0><<<g128, blk, 0, stream>>>(h, D_, Wv, D_, D_, Vb, D_, nullptr, nullptr, nullptr, nullptr, nullptr);

  // 2) Edge-wise attention scatter
  edge_attn<<<(E_ * H_ + 255) / 256, blk, 0, stream>>>(Qb, Kb, Vb, src, dst, WVb, Zb, E_);

  // 3) O projection (A = wV/(z+eps)) + bo + residual h, + BN1 stats
  gemm_wmma<1><<<g128, blk, 0, stream>>>(WVb, D_, Wo, D_, D_, X1, D_, bo, h, Zb, sum1, sq1);

  // 4) BatchNorm 1
  bn_finalize<<<1, 128, 0, stream>>>(sum1, sq1, mu1, rs1, 1.0f / (float)N_);
  bn_apply<<<(int)(ND / 256), blk, 0, stream>>>(X1, mu1, rs1, g1, be1, HX, (int)ND);

  // 5) FFN: hidden = relu(HX @ W1 + b1)
  gemm_wmma<2><<<g256, blk, 0, stream>>>(HX, D_, W1, DFF_, D_, Tb, DFF_, b1, nullptr, nullptr, nullptr, nullptr);

  // 6) y = hidden @ W2 + b2 + HX, + BN2 stats
  gemm_wmma<3><<<g128, blk, 0, stream>>>(Tb, DFF_, W2, D_, DFF_, Yb, D_, b2, HX, nullptr, sum2, sq2);

  // 7) BatchNorm 2 -> output
  bn_finalize<<<1, 128, 0, stream>>>(sum2, sq2, mu2, rs2, 1.0f / (float)N_);
  bn_apply<<<(int)(ND / 256), blk, 0, stream>>>(Yb, mu2, rs2, g2, be2, outp, (int)ND);
}
